// logTransformer_19542101197549
// MI455X (gfx1250) — matrix operand order repaired and checked by reading, not compiled, not run on hardware
//
#include <hip/hip_runtime.h>
#include <hip/hip_bf16.h>

// ---------------------------------------------------------------------------
// Model constants (from the reference)
// ---------------------------------------------------------------------------
#define Bn   32
#define SLn  256
#define DSLn 64
#define NEn  64
#define INDn 8
#define NHn  8
#define Ln   3
#define HZn  24
#define Dn   72        // NE + IND
#define DHn  576       // D * NH
#define QKN  1152      // 2 * DH

typedef unsigned short u16;
typedef __bf16  bf16;
typedef __bf16  bf16x16 __attribute__((ext_vector_type(16)));
typedef float   f32x8   __attribute__((ext_vector_type(8)));
typedef unsigned short u16x16 __attribute__((ext_vector_type(16)));
typedef unsigned short u16x8  __attribute__((ext_vector_type(8)));

__device__ __forceinline__ u16 f2bf(float f) {
    unsigned int u = __builtin_bit_cast(unsigned int, f);
    u += 0x7FFFu + ((u >> 16) & 1u);       // round-to-nearest-even
    return (u16)(u >> 16);
}

// gfx1250 async global->LDS copy of 16 bytes per lane (ASYNCcnt-tracked)
__device__ __forceinline__ void async_g2l_b128(void* lds_ptr, const void* gptr) {
    unsigned lo = (unsigned)(unsigned long long)lds_ptr;   // LDS byte offset (low 32b of flat)
    unsigned long long ga = (unsigned long long)gptr;
    asm volatile("global_load_async_to_lds_b128 %0, %1, off"
                 :: "v"(lo), "v"(ga) : "memory");
}
__device__ __forceinline__ void wait_asynccnt0() {
    asm volatile("s_wait_asynccnt 0x0" ::: "memory");
}

// ---------------------------------------------------------------------------
// Small helpers: f32 -> bf16 convert, build h0 = concat(x, zeros)
// ---------------------------------------------------------------------------
__global__ void k_cvt_bf16(const float* __restrict__ src, u16* __restrict__ dst, long n) {
    long i = (long)blockIdx.x * blockDim.x + threadIdx.x;
    if (i < n) dst[i] = f2bf(src[i]);
}

__global__ void k_build_h0(const float* __restrict__ x, float* __restrict__ h,
                           u16* __restrict__ hb, int T) {
    long i = (long)blockIdx.x * blockDim.x + threadIdx.x;
    long n = (long)Bn * T * Dn;
    if (i >= n) return;
    int c   = (int)(i % Dn);
    long bt = i / Dn;
    float v = (c < INDn) ? x[bt * INDn + c] : 0.f;
    h[i]  = v;
    hb[i] = f2bf(v);
}

// Wt[(delta*D + i) * 1152 + o] = qk_w[o, i, T-1-delta]   (bf16)
__global__ void k_prep_qkw(const float* __restrict__ w, u16* __restrict__ Wt, int T) {
    long idx = (long)blockIdx.x * blockDim.x + threadIdx.x;
    long n   = (long)QKN * Dn * T;
    if (idx >= n) return;
    int  o   = (int)(idx / ((long)Dn * T));
    long rem = idx - (long)o * Dn * T;
    int  i   = (int)(rem / T);
    int  tau = (int)(rem - (long)i * T);
    int  dlt = T - 1 - tau;
    Wt[((long)dlt * Dn + i) * QKN + o] = f2bf(w[idx]);
}

// ---------------------------------------------------------------------------
// Generic batched WMMA GEMM: C = A(bf16) x B(bf16) [+bias], fp32 accumulate.
//   block = 128 threads = 4 waves, tile = 64(M) x 64(N), K chunks of 32.
//   Wave w owns row band [w*16, w*16+16): per K-chunk it loads 1 A fragment
//   and 4 B fragments and issues 4 independent v_wmma into 4 accumulators.
//   Full tiles staged with B128 vector loads; edge tiles fall back to
//   element-wise guarded staging.  All call sites: M % 64 == 0 and 16B-
//   divisible leading dimensions; k0 % 32 == 0, n0 % 64 == 0.
// ---------------------------------------------------------------------------
#define LDS_STR 40   // padded LDS pitch (u16): 80B rows -> 16B aligned, no bank conflicts

__global__ void __launch_bounds__(128)
k_gemm(const u16* __restrict__ A, const u16* __restrict__ Bm,
       const float* __restrict__ bias,
       float* __restrict__ Cf, u16* __restrict__ Cb,
       int M, int N, int K, int lda, int ldb, int ldc,
       long sAo, long sAi, long sBo, long sBi, long sCo, long sCi,
       int inner, int transB)
{
    __shared__ u16 ldsA[64 * LDS_STR];
    __shared__ u16 ldsB[64 * LDS_STR];

    const int z  = blockIdx.z;
    const int zo = z / inner;
    const int zi = z - zo * inner;
    A  += zo * sAo + zi * sAi;
    Bm += zo * sBo + zi * sBi;
    const long cbase = zo * sCo + zi * sCi;

    const int m0 = blockIdx.y * 64;
    const int n0 = blockIdx.x * 64;

    const int lane = threadIdx.x & 31;
    const int wv   = threadIdx.x >> 5;   // 0..3, owns m-subtile [wv*16, wv*16+16)
    const int l    = lane & 15;
    const int hi   = lane >> 4;

    f32x8 acc[4] = {};

    for (int k0 = 0; k0 < K; k0 += 32) {
        const bool full = (m0 + 64 <= M) & (n0 + 64 <= N) & (k0 + 32 <= K);
        if (full) {
            // ---- A tile 64x32: 256 B128 loads ----
            for (int c = threadIdx.x; c < 256; c += 128) {
                int m = c >> 2, kc = c & 3;
                u16x8 v8 = *(const u16x8*)(A + (long)(m0 + m) * lda + k0 + kc * 8);
                *(u16x8*)&ldsA[m * LDS_STR + kc * 8] = v8;
            }
            // ---- B tile 32x64 -> ldsB[n][k] ----
            if (transB) {
                for (int c = threadIdx.x; c < 256; c += 128) {
                    int n = c >> 2, kc = c & 3;
                    u16x8 v8 = *(const u16x8*)(Bm + (long)(n0 + n) * ldb + k0 + kc * 8);
                    *(u16x8*)&ldsB[n * LDS_STR + kc * 8] = v8;
                }
            } else {
                for (int c = threadIdx.x; c < 256; c += 128) {
                    int k = c >> 3, nc = c & 7;
                    u16x8 v8 = *(const u16x8*)(Bm + (long)(k0 + k) * ldb + n0 + nc * 8);
                    #pragma unroll
                    for (int j = 0; j < 8; ++j)
                        ldsB[(nc * 8 + j) * LDS_STR + k] = v8[j];
                }
            }
        } else {
            for (int e = threadIdx.x; e < 64 * 32; e += 128) {
                int m = e >> 5, k = e & 31;
                int gm = m0 + m, gk = k0 + k;
                ldsA[m * LDS_STR + k] =
                    (gm < M && gk < K) ? A[(long)gm * lda + gk] : (u16)0;
            }
            if (transB) {
                for (int e = threadIdx.x; e < 64 * 32; e += 128) {
                    int n = e >> 5, k = e & 31;
                    int gn = n0 + n, gk = k0 + k;
                    ldsB[n * LDS_STR + k] =
                        (gn < N && gk < K) ? Bm[(long)gn * ldb + gk] : (u16)0;
                }
            } else {
                for (int e = threadIdx.x; e < 64 * 32; e += 128) {
                    int n = e & 63, k = e >> 6;
                    int gn = n0 + n, gk = k0 + k;
                    ldsB[n * LDS_STR + k] =
                        (gn < N && gk < K) ? Bm[(long)gk * ldb + gn] : (u16)0;
                }
            }
        }
        __syncthreads();

        // ---- 1 A fragment + 4 B fragments -> 4 independent WMMAs ----
        u16x16 ap;
        #pragma unroll
        for (int v = 0; v < 8; ++v) {
            int ka = ((v < 4) ? 0 : 16) + hi * 8 + 2 * (v & 3);
            ap[2 * v]     = ldsA[(wv * 16 + l) * LDS_STR + ka];
            ap[2 * v + 1] = ldsA[(wv * 16 + l) * LDS_STR + ka + 1];
        }
        bf16x16 af = __builtin_bit_cast(bf16x16, ap);
        #pragma unroll
        for (int s = 0; s < 4; ++s) {
            u16x16 bp;
            #pragma unroll
            for (int v = 0; v < 8; ++v) {
                int kb = ((v < 4) ? 0 : 16) + hi * 8 + 2 * (v & 3);
                bp[2 * v]     = ldsB[(s * 16 + l) * LDS_STR + kb];
                bp[2 * v + 1] = ldsB[(s * 16 + l) * LDS_STR + kb + 1];
            }
            bf16x16 bfm = __builtin_bit_cast(bf16x16, bp);
            acc[s] = __builtin_amdgcn_wmma_f32_16x16x32_bf16(
                         false, af, false, bfm, (short)0, acc[s], false, false);
        }
        __syncthreads();
    }

    // ---- epilogue: C VGPR r -> (M = wv*16 + r + 8*hi, N = s*16 + l) ----
    #pragma unroll
    for (int s = 0; s < 4; ++s) {
        int gn = n0 + s * 16 + l;
        if (gn < N) {
            float bv = bias ? bias[gn] : 0.f;
            #pragma unroll
            for (int r = 0; r < 8; ++r) {
                int gm = m0 + wv * 16 + r + 8 * hi;
                if (gm < M) {
                    float v = acc[s][r] + bv;
                    long  o = cbase + (long)gm * ldc + gn;
                    if (Cf) Cf[o] = v;
                    if (Cb) Cb[o] = f2bf(v);
                }
            }
        }
    }
}

// ---------------------------------------------------------------------------
// Causal conv (qk) as a Toeplitz WMMA GEMM.
//   k = delta*72 + i,  A[t][k] = h[b, t-delta, i] (0 if t-delta < 0),
//   B = Wt (pre-transposed bf16).  Full per-batch history resides in LDS,
//   loaded via GLOBAL_LOAD_ASYNC_TO_LDS_B128; weight tiles are ASYNCcnt-
//   tracked async copies ([k][n] layout keeps LDS writes contiguous).
//   Tile = 64(t) x 64(o): wave w owns t-band, 4 accumulators over o.
//   grid = (1152/64, T/64, B), block = 128 (4 waves).
// ---------------------------------------------------------------------------
__global__ void __launch_bounds__(128)
k_qkconv(const u16* __restrict__ hb, const u16* __restrict__ Wt,
         const float* __restrict__ qkb, u16* __restrict__ qkOut, int T)
{
    __shared__ u16 hbuf[SLn * Dn];          // 36 KB max, fits 320 KB WGP LDS
    __shared__ u16 ldsBkn[32 * 64];         // [k][n], 4 KB, rows 128B (16B aligned)

    const int b  = blockIdx.z;
    const int t0 = blockIdx.y * 64;
    const int o0 = blockIdx.x * 64;

    // async-load whole history h[b] (bf16) into LDS once: 16B per lane issue
    const u16* hsrc = hb + (long)b * T * Dn;
    {
        const int nchunk = (T * Dn) >> 3;   // 2304 (t1) / 576 (t2)
        for (int c = threadIdx.x; c < nchunk; c += 128)
            async_g2l_b128(&hbuf[c * 8], hsrc + (long)c * 8);
        wait_asynccnt0();
    }
    __syncthreads();

    const int lane = threadIdx.x & 31;
    const int wv   = threadIdx.x >> 5;
    const int l    = lane & 15;
    const int hi   = lane >> 4;
    const int trow = t0 + wv * 16 + l;      // this wave's A row

    f32x8 acc[4] = {};
    const int Kend = (t0 + 64) * Dn;        // exact multiple of 32

    for (int k0 = 0; k0 < Kend; k0 += 32) {
        // async stage weight tile 32x64 -> ldsBkn[k][n] (contiguous 16B chunks)
        for (int c = threadIdx.x; c < 256; c += 128) {
            int k  = c >> 3;                // 0..31
            int nc = c & 7;                 // chunk of 8 cols
            async_g2l_b128(&ldsBkn[k * 64 + nc * 8],
                           Wt + (long)(k0 + k) * QKN + o0 + nc * 8);
        }
        wait_asynccnt0();
        __syncthreads();

        // Toeplitz A fragment from LDS history (causal zeros via s >= 0)
        u16x16 ap;
        #pragma unroll
        for (int v = 0; v < 8; ++v) {
            int ka = k0 + ((v < 4) ? 0 : 16) + hi * 8 + 2 * (v & 3);
            #pragma unroll
            for (int p = 0; p < 2; ++p) {
                int kk  = ka + p;
                int dlt = kk / Dn;
                int ii  = kk - dlt * Dn;
                int s   = trow - dlt;
                ap[2 * v + p] = (s >= 0) ? hbuf[s * Dn + ii] : (u16)0;
            }
        }
        bf16x16 af = __builtin_bit_cast(bf16x16, ap);
        #pragma unroll
        for (int s = 0; s < 4; ++s) {
            u16x16 bp;
            #pragma unroll
            for (int v = 0; v < 8; ++v) {
                int kb = ((v < 4) ? 0 : 16) + hi * 8 + 2 * (v & 3);
                bp[2 * v]     = ldsBkn[kb * 64 + s * 16 + l];
                bp[2 * v + 1] = ldsBkn[(kb + 1) * 64 + s * 16 + l];
            }
            bf16x16 bfm = __builtin_bit_cast(bf16x16, bp);
            acc[s] = __builtin_amdgcn_wmma_f32_16x16x32_bf16(
                         false, af, false, bfm, (short)0, acc[s], false, false);
        }
        __syncthreads();
    }

    #pragma unroll
    for (int s = 0; s < 4; ++s) {
        int gn = o0 + s * 16 + l;           // always < 1152
        float bv = qkb[gn];
        #pragma unroll
        for (int r = 0; r < 8; ++r) {
            int gm = t0 + wv * 16 + r + 8 * hi;   // always < T
            qkOut[((long)b * T + gm) * QKN + gn] = f2bf(acc[s][r] + bv);
        }
    }
}

// ---------------------------------------------------------------------------
// Causal softmax: one block per (b,h,q) row; writes bf16 attention weights.
// ---------------------------------------------------------------------------
__global__ void k_softmax(const float* __restrict__ S, u16* __restrict__ Att, int T) {
    const long row = blockIdx.x;
    const int  q   = (int)(row % T);
    const int  len = q + 1;
    const float* s = S + row * T;
    u16* a = Att + row * T;
    __shared__ float red[128];
    const int tid = threadIdx.x;

    float mx = -1e30f;
    for (int k = tid; k < len; k += 128) mx = fmaxf(mx, s[k]);
    red[tid] = mx; __syncthreads();
    for (int st = 64; st > 0; st >>= 1) {
        if (tid < st) red[tid] = fmaxf(red[tid], red[tid + st]);
        __syncthreads();
    }
    mx = red[0]; __syncthreads();

    float sum = 0.f;
    for (int k = tid; k < len; k += 128) sum += __expf(s[k] - mx);
    red[tid] = sum; __syncthreads();
    for (int st = 64; st > 0; st >>= 1) {
        if (tid < st) red[tid] += red[tid + st];
        __syncthreads();
    }
    sum = red[0]; __syncthreads();
    float inv = 1.f / sum;

    for (int k = tid; k < T; k += 128)
        a[k] = (k < len) ? f2bf(__expf(s[k] - mx) * inv) : (u16)0;
}

// ---------------------------------------------------------------------------
// h = LayerNorm(h + a); writes f32 and bf16 copies. One block per row (D=72).
// ---------------------------------------------------------------------------
__global__ void k_res_ln(float* __restrict__ h, const float* __restrict__ a,
                         u16* __restrict__ hbf,
                         const float* __restrict__ g, const float* __restrict__ bb)
{
    const long row = blockIdx.x;
    const int  c   = threadIdx.x;
    __shared__ float red[128];

    float v = 0.f;
    if (c < Dn) v = h[row * Dn + c] + a[row * Dn + c];
    red[c] = (c < Dn) ? v : 0.f;
    __syncthreads();
    for (int st = 64; st > 0; st >>= 1) {
        if (c < st) red[c] += red[c + st];
        __syncthreads();
    }
    float mu = red[0] * (1.f / Dn);
    __syncthreads();

    float d = (c < Dn) ? (v - mu) : 0.f;
    red[c] = d * d;
    __syncthreads();
    for (int st = 64; st > 0; st >>= 1) {
        if (c < st) red[c] += red[c + st];
        __syncthreads();
    }
    float var = red[0] * (1.f / Dn);

    if (c < Dn) {
        float o = g[c] * (v - mu) * rsqrtf(var + 1e-5f) + bb[c];
        h[row * Dn + c]   = o;
        hbf[row * Dn + c] = f2bf(o);
    }
}

// ---------------------------------------------------------------------------
// Sigma conv: out[b,hz,j] = bias[hz] + sum_{c,w} h[b,c,j+w] * W[hz,c,w]
// writes into fused feature buffer f[(b*24+hz)*128 + off + j]
// ---------------------------------------------------------------------------
__global__ void k_sigma(const float* __restrict__ h, const float* __restrict__ W,
                        const float* __restrict__ bias, float* __restrict__ fbuf,
                        int T, int off)
{
    const int bhz = blockIdx.x;           // b*24 + hz
    const int b   = bhz / HZn;
    const int hz  = bhz - b * HZn;
    const int j   = threadIdx.x;          // 0..63
    float acc = bias[hz];
    for (int c = 0; c < T; ++c) {
        const float* hr = h + ((long)b * T + c) * Dn + j;
        const float* wr = W + ((long)hz * T + c) * (INDn + 1);
        #pragma unroll
        for (int w = 0; w < INDn + 1; ++w) acc += hr[w] * wr[w];
    }
    fbuf[(long)bhz * 128 + off + j] = acc;
}

// ---------------------------------------------------------------------------
// MLP head: tanh(128->128) -> tanh(128->64) -> tanh(64->64) -> 64->1
// one block (128 threads) per (b,hz) row.
// ---------------------------------------------------------------------------
__global__ void k_head(const float* __restrict__ fbuf,
                       const float* __restrict__ w1, const float* __restrict__ b1,
                       const float* __restrict__ w2, const float* __restrict__ b2,
                       const float* __restrict__ w3, const float* __restrict__ b3,
                       const float* __restrict__ w4, const float* __restrict__ b4,
                       float* __restrict__ out)
{
    const long row = blockIdx.x;
    const int  t   = threadIdx.x;
    __shared__ float f0[128], f1[128];

    f0[t] = fbuf[row * 128 + t];
    __syncthreads();

    float acc = b1[t];
    for (int i = 0; i < 128; ++i) acc += f0[i] * w1[i * 128 + t];
    f1[t] = tanhf(acc);
    __syncthreads();

    if (t < 64) {
        acc = b2[t];
        for (int i = 0; i < 128; ++i) acc += f1[i] * w2[i * 64 + t];
        f0[t] = tanhf(acc);
    }
    __syncthreads();

    if (t < 64) {
        acc = b3[t];
        for (int i = 0; i < 64; ++i) acc += f0[i] * w3[i * 64 + t];
        f1[t] = tanhf(acc);
    }
    __syncthreads();

    if (t == 0) {
        acc = b4[0];
        for (int i = 0; i < 64; ++i) acc += f1[i] * w4[i];
        out[row] = acc;
    }
}

// ---------------------------------------------------------------------------
// Orchestration
// ---------------------------------------------------------------------------
extern "C" void kernel_launch(void* const* d_in, const int* in_sizes, int n_in,
                              void* d_out, int out_size, void* d_ws, size_t ws_size,
                              hipStream_t stream)
{
    (void)in_sizes; (void)n_in; (void)out_size; (void)ws_size;

    char* ws = (char*)d_ws;
    size_t off = 0;
    auto alloc = [&](size_t bytes) -> char* {
        char* p = ws + off;
        off += (bytes + 255) & ~(size_t)255;
        return p;
    };

    float* hF   = (float*)alloc((size_t)Bn * SLn * Dn * 4);
    u16*   hB   = (u16*)  alloc((size_t)Bn * SLn * Dn * 2);
    float* aF   = (float*)alloc((size_t)Bn * SLn * Dn * 4);
    u16*   vB   = (u16*)  alloc((size_t)Bn * SLn * DHn * 2);
    u16*   qkB  = (u16*)  alloc((size_t)Bn * SLn * QKN * 2);
    u16*   atnB = (u16*)  alloc((size_t)Bn * SLn * DHn * 2);
    float* sc   = (float*)alloc((size_t)Bn * NHn * SLn * SLn * 4);
    u16*   attB = (u16*)  alloc((size_t)Bn * NHn * SLn * SLn * 2);
    u16*   WtB  = (u16*)  alloc((size_t)SLn * Dn * QKN * 2);
    u16*   vWB  = (u16*)  alloc((size_t)Dn * DHn * 2);
    u16*   pWB  = (u16*)  alloc((size_t)DHn * Dn * 2);
    float* fbuf = (float*)alloc((size_t)Bn * HZn * 128 * 4);

    auto gemm = [&](const u16* A, const u16* Bm, const float* bias,
                    float* Cf, u16* Cb,
                    int M, int N, int K, int lda, int ldb, int ldc,
                    long sAo, long sAi, long sBo, long sBi, long sCo, long sCi,
                    int inner, int batches, int transB) {
        dim3 g((N + 63) / 64, (M + 63) / 64, batches);
        k_gemm<<<g, 128, 0, stream>>>(A, Bm, bias, Cf, Cb, M, N, K,
                                      lda, ldb, ldc,
                                      sAo, sAi, sBo, sBi, sCo, sCi,
                                      inner, transB);
    };

    for (int tr = 0; tr < 2; ++tr) {
        const int T    = tr ? DSLn : SLn;
        const int base = tr ? 10 : 2;
        const float* xin = (const float*)d_in[tr];

        // h0 = concat(x, zeros)
        {
            long n = (long)Bn * T * Dn;
            k_build_h0<<<(unsigned)((n + 255) / 256), 256, 0, stream>>>(xin, hF, hB, T);
        }

        for (int l = 0; l < Ln; ++l) {
            const float* qkw = (const float*)d_in[base + 0] + (long)l * QKN * Dn * T;
            const float* qkb = (const float*)d_in[base + 1] + (long)l * QKN;
            const float* vw  = (const float*)d_in[base + 2] + (long)l * Dn * DHn;
            const float* vb  = (const float*)d_in[base + 3] + (long)l * DHn;
            const float* pw  = (const float*)d_in[base + 4] + (long)l * DHn * Dn;
            const float* pb  = (const float*)d_in[base + 5] + (long)l * Dn;
            const float* lng = (const float*)d_in[base + 6] + (long)l * Dn;
            const float* lnb = (const float*)d_in[base + 7] + (long)l * Dn;

            // weight prep (bf16)
            {
                long n = (long)QKN * Dn * T;
                k_prep_qkw<<<(unsigned)((n + 255) / 256), 256, 0, stream>>>(qkw, WtB, T);
                k_cvt_bf16<<<(Dn * DHn + 255) / 256, 256, 0, stream>>>(vw, vWB, Dn * DHn);
                k_cvt_bf16<<<(DHn * Dn + 255) / 256, 256, 0, stream>>>(pw, pWB, DHn * Dn);
            }

            // V = h @ v_w + v_b   (bf16 out)
            gemm(hB, vWB, vb, (float*)nullptr, vB,
                 Bn * T, DHn, Dn, Dn, DHn, DHn,
                 0, 0, 0, 0, 0, 0, 1, 1, 0);

            // QK causal conv (Toeplitz WMMA), fused bias, bf16 out
            {
                dim3 g(QKN / 64, T / 64, Bn);
                k_qkconv<<<g, 128, 0, stream>>>(hB, WtB, qkb, qkB, T);
            }

            // scores = Q @ K^T  per (b,h)
            gemm(qkB, qkB + DHn, nullptr, sc, (u16*)nullptr,
                 T, T, Dn, QKN, QKN, T,
                 (long)T * QKN, Dn, (long)T * QKN, Dn,
                 (long)NHn * T * T, (long)T * T,
                 NHn, Bn * NHn, 1);

            // causal softmax -> bf16 attention weights
            k_softmax<<<(unsigned)(Bn * NHn * T), 128, 0, stream>>>(sc, attB, T);

            // out = att @ V  per (b,h), bf16 out (B,T,DH) layout
            gemm(attB, vB, nullptr, (float*)nullptr, atnB,
                 T, Dn, T, T, DHn, DHn,
                 (long)NHn * T * T, (long)T * T,
                 (long)T * DHn, Dn,
                 (long)T * DHn, Dn,
                 NHn, Bn * NHn, 0);

            // a = out @ p_w + p_b  (f32)
            gemm(atnB, pWB, pb, aF, (u16*)nullptr,
                 Bn * T, Dn, DHn, DHn, Dn, Dn,
                 0, 0, 0, 0, 0, 0, 1, 1, 0);

            // h = LN(h + a)
            k_res_ln<<<(unsigned)(Bn * T), 128, 0, stream>>>(hF, aF, hB, lng, lnb);
        }

        // sigma conv -> fused feature buffer halves
        {
            const float* sw = (const float*)d_in[tr ? 20 : 18];
            const float* sb = (const float*)d_in[tr ? 21 : 19];
            k_sigma<<<(unsigned)(Bn * HZn), 64, 0, stream>>>(hF, sw, sb, fbuf, T, tr ? 64 : 0);
        }
    }

    // MLP head
    k_head<<<(unsigned)(Bn * HZn), 128, 0, stream>>>(
        fbuf,
        (const float*)d_in[22], (const float*)d_in[23],
        (const float*)d_in[24], (const float*)d_in[25],
        (const float*)d_in[26], (const float*)d_in[27],
        (const float*)d_in[28], (const float*)d_in[29],
        (float*)d_out);
}
